// SwinTransformerBlock_49100066128649
// MI455X (gfx1250) — matrix operand order repaired and checked by reading, not compiled, not run on hardware
//
#include <hip/hip_runtime.h>
#include <hip/hip_bf16.h>
#include <math.h>

typedef __attribute__((ext_vector_type(8)))  _Float16 v8h;
typedef __attribute__((ext_vector_type(16))) _Float16 v16h;
typedef __attribute__((ext_vector_type(8)))  float    v8f;

namespace {
constexpr int kB     = 32;
constexpr int kH     = 56;
constexpr int kDIM   = 384;
constexpr int kHEADS = 12;
constexpr int kWIN   = 7;
constexpr int kSHIFT = 3;
constexpr int kHD    = 32;              // head dim
constexpr int kNW    = 64;              // 8*8 windows
constexpr int kL     = kH * kH + 1;     // 3137
constexpr int kHID   = 4 * kDIM;        // 1536
constexpr int kM     = kB * kL;         // 100384 rows
constexpr float kSCALE = 0.17677669529663687f;  // 1/sqrt(32)
}

// ---------------------------------------------------------------- wmma core

__device__ __forceinline__ v8f wmma_f16(v16h a, v16h b, v8f c) {
#if defined(__gfx1250__)
  return __builtin_amdgcn_wmma_f32_16x16x32_f16(false, a, false, b, (short)0, c,
                                                false, false);
#else
  return c;  // host-pass stub
#endif
}

__device__ __forceinline__ v8f zero8f() {
  v8f z;
#pragma unroll
  for (int e = 0; e < 8; ++e) z[e] = 0.f;
  return z;
}

__device__ __forceinline__ v8h zero8h() {
  v8h z;
#pragma unroll
  for (int e = 0; e < 8; ++e) z[e] = (_Float16)0.f;
  return z;
}

// A-fragment: 16x32 f16, row-major source with leading dim lda.
// lanes 0-15: row m, K = k0..k0+15
// lanes 16-31: row m, K = k0+16..k0+31
__device__ __forceinline__ v16h ld_a_frag(const _Float16* A, int lda, int row0,
                                          int k0, int maxrow) {
  int lane = threadIdx.x & 31;
  int half = lane >> 4;
  int r = row0 + (lane & 15);
  if (r > maxrow) r = maxrow;
  const _Float16* p = A + (size_t)r * lda + k0 + half * 16;
  v8h lo = *(const v8h*)p;
  v8h hi = *(const v8h*)(p + 8);
  v16h f;
#pragma unroll
  for (int i = 0; i < 8; ++i) { f[i] = lo[i]; f[i + 8] = hi[i]; }
  return f;
}

// B-fragment: B[k,n] = Wt[n,k] (Wt row-major N x K with leading dim ldw).
// lanes 0-15: col n, K = k0..k0+15 ; lanes 16-31: col n, K = k0+16..k0+31
__device__ __forceinline__ v16h ld_b_frag(const _Float16* Wt, int ldw, int col0,
                                          int k0) {
  int lane = threadIdx.x & 31;
  int half = lane >> 4;
  int n = col0 + (lane & 15);
  const _Float16* p = Wt + (size_t)n * ldw + k0 + half * 16;
  v8h lo = *(const v8h*)p;
  v8h hi = *(const v8h*)(p + 8);
  v16h f;
#pragma unroll
  for (int i = 0; i < 8; ++i) { f[i] = lo[i]; f[i + 8] = hi[i]; }
  return f;
}

// B-fragment from a K-major buffer vbuf[64][32]: B[k,n] = vbuf[k*32+n]
__device__ __forceinline__ v16h ld_bcol_frag(const _Float16* vbuf, int k0, int n0) {
  int lane = threadIdx.x & 31;
  int half = lane >> 4;
  int n = n0 + (lane & 15);
  int kb = k0 + half * 16;
  v16h f;
#pragma unroll
  for (int j = 0; j < 16; ++j) f[j] = vbuf[(kb + j) * 32 + n];
  return f;
}

// ---------------------------------------------------------------- small kernels

__global__ __launch_bounds__(256) void cast_w_kernel(const float* __restrict__ src,
                                                     _Float16* __restrict__ dst,
                                                     int n) {
  int i = blockIdx.x * 256 + threadIdx.x;
  if (i < n) dst[i] = (_Float16)src[i];
}

// LayerNorm over rows of length 384, one wave32 per row, f16 output.
__global__ __launch_bounds__(256) void ln_kernel(const float* __restrict__ x,
                                                 const float* __restrict__ g,
                                                 const float* __restrict__ bb,
                                                 _Float16* __restrict__ o,
                                                 int rows) {
  int row = blockIdx.x * 8 + (threadIdx.x >> 5);
  int lane = threadIdx.x & 31;
  if (row >= rows) return;
  const float* xr = x + (size_t)row * kDIM;
  float v[12];
  float s = 0.f;
#pragma unroll
  for (int i = 0; i < 12; ++i) { v[i] = xr[lane + i * 32]; s += v[i]; }
#pragma unroll
  for (int off = 16; off > 0; off >>= 1) s += __shfl_xor(s, off, 32);
  float mean = s * (1.0f / kDIM);
  float var = 0.f;
#pragma unroll
  for (int i = 0; i < 12; ++i) { float d = v[i] - mean; var += d * d; }
#pragma unroll
  for (int off = 16; off > 0; off >>= 1) var += __shfl_xor(var, off, 32);
  float rstd = rsqrtf(var * (1.0f / kDIM) + 1e-5f);
  _Float16* orow = o + (size_t)row * kDIM;
#pragma unroll
  for (int i = 0; i < 12; ++i) {
    int c = lane + i * 32;
    orow[c] = (_Float16)((v[i] - mean) * rstd * g[c] + bb[c]);
  }
}

// ---------------------------------------------------------------- GEMM
// C[M,N] = A[M,K] @ W[N,K]^T (+bias) (+residual / gelu), f16 inputs f32 acc.
// Block: 256 thr = 8 waves, tile 128x128; wave = 32x64 (2 M-tiles x 4 N-tiles).
// mode 0: f16 out = acc+bias     (qkv)
// mode 1: f32 out = acc+bias+res (proj -> x1, fc2 -> out)
// mode 2: f16 out = gelu(acc+bias) (fc1)
__global__ __launch_bounds__(256) void gemm_kernel(
    const _Float16* __restrict__ A, const _Float16* __restrict__ W,
    const float* __restrict__ bias, void* __restrict__ outp,
    const float* __restrict__ res, int M, int N, int K, int mode) {
  int wv = threadIdx.x >> 5;
  int lane = threadIdx.x & 31;
  int half = lane >> 4;
  int cn = lane & 15;
  int row0 = blockIdx.x * 128 + (wv >> 1) * 32;
  int col0 = blockIdx.y * 128 + (wv & 1) * 64;

  v8f acc[2][4];
#pragma unroll
  for (int i = 0; i < 2; ++i)
#pragma unroll
    for (int j = 0; j < 4; ++j) acc[i][j] = zero8f();

  for (int k0 = 0; k0 < K; k0 += 32) {
    v16h a0 = ld_a_frag(A, K, row0, k0, M - 1);
    v16h a1 = ld_a_frag(A, K, row0 + 16, k0, M - 1);
#pragma unroll
    for (int j = 0; j < 4; ++j) {
      v16h bf = ld_b_frag(W, K, col0 + j * 16, k0);
      acc[0][j] = wmma_f16(a0, bf, acc[0][j]);
      acc[1][j] = wmma_f16(a1, bf, acc[1][j]);
    }
  }

#pragma unroll
  for (int i = 0; i < 2; ++i) {
#pragma unroll
    for (int j = 0; j < 4; ++j) {
      int col = col0 + j * 16 + cn;
      float bv = bias[col];
#pragma unroll
      for (int e = 0; e < 8; ++e) {
        int r = row0 + i * 16 + half * 8 + e;
        if (r < M) {
          float v = acc[i][j][e] + bv;
          size_t idx = (size_t)r * N + col;
          if (mode == 1) {
            ((float*)outp)[idx] = v + res[idx];
          } else if (mode == 2) {
            float gg = 0.5f * v * (1.0f + erff(v * 0.7071067811865475f));
            ((_Float16*)outp)[idx] = (_Float16)gg;
          } else {
            ((_Float16*)outp)[idx] = (_Float16)v;
          }
        }
      }
    }
  }
}

// ---------------------------------------------------------------- attention

// window pos p (0..48) in window (wh,ww) -> token index (1 + unshifted h*56+w)
__device__ __forceinline__ int win_token(int wh, int ww, int p) {
  int ph = p / kWIN, pw = p % kWIN;
  int hh = wh * kWIN + ph, wc = ww * kWIN + pw;
  int sh = (hh + kSHIFT) % kH, sw = (wc + kSHIFT) % kH;
  return 1 + sh * kH + sw;
}

// region id of a shifted-image coordinate (for the shift attention mask)
__device__ __forceinline__ int region_id(int hh, int wc) {
  int rh = (hh < kH - kWIN) ? 0 : ((hh < kH - kSHIFT) ? 1 : 2);
  int rw = (wc < kH - kWIN) ? 0 : ((wc < kH - kSHIFT) ? 1 : 2);
  return rh * 3 + rw;
}

// one block = one (window, head). 128 threads = 4 waves.
__global__ __launch_bounds__(128) void attn_win_kernel(
    const _Float16* __restrict__ qkv, const float* __restrict__ rel_tab,
    _Float16* __restrict__ outp) {
  __shared__ __align__(16) _Float16 qb[64 * 32];
  __shared__ __align__(16) _Float16 kb[64 * 32];
  __shared__ __align__(16) _Float16 vb[64 * 32];
  __shared__ __align__(16) float    sc[64 * 64];
  __shared__ __align__(16) _Float16 pa[64 * 64];

  int wb = blockIdx.x;
  int h = blockIdx.y;
  int b = wb >> 6;
  int w = wb & 63;
  int wh = w >> 3, ww = w & 7;
  int tid = threadIdx.x;

  // stage q (rows 0..48), k/v (rows 0..49, row 0 = cls) into LDS, zero-pad to 64
  int part = tid & 3;
  for (int r = tid >> 2; r < 64; r += 32) {
    v8h qv = zero8h(), kv = zero8h(), vv = zero8h();
    if (r < 49) {
      int tok = win_token(wh, ww, r);
      const _Float16* p =
          qkv + ((size_t)b * kL + tok) * (3 * kDIM) + h * kHD + part * 8;
      qv = *(const v8h*)p;
    }
    if (r < 50) {
      int tok = (r == 0) ? 0 : win_token(wh, ww, r - 1);
      const _Float16* p =
          qkv + ((size_t)b * kL + tok) * (3 * kDIM) + h * kHD + part * 8;
      kv = *(const v8h*)(p + kDIM);
      vv = *(const v8h*)(p + 2 * kDIM);
    }
    *(v8h*)(qb + r * 32 + part * 8) = qv;
    *(v8h*)(kb + r * 32 + part * 8) = kv;
    *(v8h*)(vb + r * 32 + part * 8) = vv;
  }
  __syncthreads();

  // scores = q @ k^T  (K = 32 = HEAD_DIM, one wmma per 16x16 tile)
  int wv = tid >> 5, lane = tid & 31, half = lane >> 4, cn = lane & 15;
  int m0 = wv * 16;
  {
    v16h af = ld_a_frag(qb, 32, m0, 0, 63);
#pragma unroll
    for (int j = 0; j < 4; ++j) {
      v16h bf = ld_b_frag(kb, 32, j * 16, 0);
      v8f s = wmma_f16(af, bf, zero8f());
#pragma unroll
      for (int e = 0; e < 8; ++e)
        sc[(m0 + half * 8 + e) * 64 + j * 16 + cn] = s[e];
    }
  }
  __syncthreads();

  // softmax with scale, relative bias and shift mask; one thread per row
  if (tid < 64) {
    int r = tid;
    if (r < 49) {
      int qh = r / kWIN, qw = r % kWIN;
      int qreg = region_id(wh * kWIN + qh, ww * kWIN + qw);
      float mx = -1e30f;
      for (int n = 0; n < 50; ++n) {
        float v = sc[r * 64 + n] * kSCALE;
        if (n > 0) {
          int p = n - 1;
          int kh = p / kWIN, kw = p % kWIN;
          int ridx = (qh - kh + kWIN - 1) * (2 * kWIN - 1) + (qw - kw + kWIN - 1);
          v += rel_tab[ridx * kHEADS + h];
          if (region_id(wh * kWIN + kh, ww * kWIN + kw) != qreg) v -= 100.f;
        }
        sc[r * 64 + n] = v;
        mx = fmaxf(mx, v);
      }
      float sum = 0.f;
      for (int n = 0; n < 50; ++n) {
        float e = expf(sc[r * 64 + n] - mx);
        sc[r * 64 + n] = e;
        sum += e;
      }
      float rs = 1.0f / sum;
      for (int n = 0; n < 50; ++n) pa[r * 64 + n] = (_Float16)(sc[r * 64 + n] * rs);
      for (int n = 50; n < 64; ++n) pa[r * 64 + n] = (_Float16)0.f;
    } else {
      for (int n = 0; n < 64; ++n) pa[r * 64 + n] = (_Float16)0.f;
    }
  }
  __syncthreads();

  // out = attn @ v : M=64(q rows) x N=32(head dim), K=64 (2 wmma K-steps)
  v8f o0 = zero8f(), o1 = zero8f();
  for (int k0 = 0; k0 < 64; k0 += 32) {
    v16h af = ld_a_frag(pa, 64, m0, k0, 63);
    v16h b0 = ld_bcol_frag(vb, k0, 0);
    v16h b1 = ld_bcol_frag(vb, k0, 16);
    o0 = wmma_f16(af, b0, o0);
    o1 = wmma_f16(af, b1, o1);
  }
#pragma unroll
  for (int e = 0; e < 8; ++e) {
    int r = m0 + half * 8 + e;
    if (r < 49) {
      int tok = win_token(wh, ww, r);
      _Float16* op = outp + ((size_t)b * kL + tok) * kDIM + h * kHD;
      op[cn] = (_Float16)o0[e];
      op[16 + cn] = (_Float16)o1[e];
    }
  }
}

// cls-token attention: one block per (b, head), full-sequence softmax (L=3137)
__global__ __launch_bounds__(256) void attn_cls_kernel(
    const _Float16* __restrict__ qkv, _Float16* __restrict__ outp) {
  __shared__ float logits[kL];
  __shared__ float red[256];
  __shared__ float part[256];
  int b = blockIdx.x, h = blockIdx.y;
  int tid = threadIdx.x;
  const _Float16* base = qkv + (size_t)b * kL * (3 * kDIM);

  float q[kHD];
#pragma unroll
  for (int d = 0; d < kHD; ++d) q[d] = (float)base[h * kHD + d];

  float lmax = -1e30f;
  for (int t = tid; t < kL; t += 256) {
    const _Float16* kp = base + (size_t)t * (3 * kDIM) + kDIM + h * kHD;
    float s = 0.f;
#pragma unroll
    for (int d = 0; d < kHD; ++d) s += q[d] * (float)kp[d];
    s *= kSCALE;
    logits[t] = s;
    lmax = fmaxf(lmax, s);
  }
  red[tid] = lmax;
  __syncthreads();
  for (int s = 128; s > 0; s >>= 1) {
    if (tid < s) red[tid] = fmaxf(red[tid], red[tid + s]);
    __syncthreads();
  }
  float mx = red[0];
  __syncthreads();

  float lsum = 0.f;
  for (int t = tid; t < kL; t += 256) {
    float e = expf(logits[t] - mx);
    logits[t] = e;
    lsum += e;
  }
  red[tid] = lsum;
  __syncthreads();
  for (int s = 128; s > 0; s >>= 1) {
    if (tid < s) red[tid] += red[tid + s];
    __syncthreads();
  }
  float rsum = 1.0f / red[0];

  int d = tid & 31, c = tid >> 5;
  float acc = 0.f;
  for (int t = c; t < kL; t += 8) {
    const _Float16* vp = base + (size_t)t * (3 * kDIM) + 2 * kDIM + h * kHD;
    acc += logits[t] * (float)vp[d];
  }
  part[tid] = acc;
  __syncthreads();
  if (tid < 32) {
    float tot = 0.f;
#pragma unroll
    for (int cc = 0; cc < 8; ++cc) tot += part[cc * 32 + tid];
    outp[(size_t)b * kL * kDIM + h * kHD + tid] = (_Float16)(tot * rsum);
  }
}

// ---------------------------------------------------------------- launch

extern "C" void kernel_launch(void* const* d_in, const int* in_sizes, int n_in,
                              void* d_out, int out_size, void* d_ws,
                              size_t ws_size, hipStream_t stream) {
  (void)in_sizes; (void)n_in; (void)out_size; (void)ws_size;
  const float* x       = (const float*)d_in[0];
  const float* n1g     = (const float*)d_in[1];
  const float* n1b     = (const float*)d_in[2];
  const float* qkv_w   = (const float*)d_in[3];
  const float* qkv_b   = (const float*)d_in[4];
  const float* rel_tab = (const float*)d_in[5];
  const float* proj_w  = (const float*)d_in[6];
  const float* proj_b  = (const float*)d_in[7];
  const float* n2g     = (const float*)d_in[8];
  const float* n2b     = (const float*)d_in[9];
  const float* fc1_w   = (const float*)d_in[10];
  const float* fc1_b   = (const float*)d_in[11];
  const float* fc2_w   = (const float*)d_in[12];
  const float* fc2_b   = (const float*)d_in[13];
  float* out = (float*)d_out;

  char* wsp = (char*)d_ws;
  size_t off = 0;
  auto alloc = [&](size_t bytes) -> void* {
    off = (off + 255) & ~(size_t)255;
    void* p = wsp + off;
    off += bytes;
    return p;
  };

  _Float16* wqkv16 = (_Float16*)alloc((size_t)3 * kDIM * kDIM * 2);
  _Float16* wproj16 = (_Float16*)alloc((size_t)kDIM * kDIM * 2);
  _Float16* wfc1_16 = (_Float16*)alloc((size_t)kHID * kDIM * 2);
  _Float16* wfc2_16 = (_Float16*)alloc((size_t)kDIM * kHID * 2);
  _Float16* h16    = (_Float16*)alloc((size_t)kM * kDIM * 2);  // LN1 out, later LN2 out
  _Float16* big    = (_Float16*)alloc((size_t)kM * kHID * 2);  // qkv16, later fc1 out
  _Float16* attn16 = (_Float16*)alloc((size_t)kM * kDIM * 2);  // attention output
  float*    x1     = (float*)alloc((size_t)kM * kDIM * 4);     // first residual sum

  // 1. weights -> f16
  {
    int n = 3 * kDIM * kDIM;
    cast_w_kernel<<<(n + 255) / 256, 256, 0, stream>>>(qkv_w, wqkv16, n);
    n = kDIM * kDIM;
    cast_w_kernel<<<(n + 255) / 256, 256, 0, stream>>>(proj_w, wproj16, n);
    n = kHID * kDIM;
    cast_w_kernel<<<(n + 255) / 256, 256, 0, stream>>>(fc1_w, wfc1_16, n);
    n = kDIM * kHID;
    cast_w_kernel<<<(n + 255) / 256, 256, 0, stream>>>(fc2_w, wfc2_16, n);
  }

  int mblocks = (kM + 127) / 128;  // 785

  // 2. LN1
  ln_kernel<<<(kM + 7) / 8, 256, 0, stream>>>(x, n1g, n1b, h16, kM);

  // 3. qkv = LN1 @ Wqkv^T + b  (f16)
  gemm_kernel<<<dim3(mblocks, (3 * kDIM) / 128), 256, 0, stream>>>(
      h16, wqkv16, qkv_b, big, nullptr, kM, 3 * kDIM, kDIM, 0);

  // 4. window attention + cls attention -> attn16
  attn_win_kernel<<<dim3(kB * kNW, kHEADS), 128, 0, stream>>>(big, rel_tab,
                                                              attn16);
  attn_cls_kernel<<<dim3(kB, kHEADS), 256, 0, stream>>>(big, attn16);

  // 5. x1 = x + attn @ Wproj^T + b  (f32)
  gemm_kernel<<<dim3(mblocks, kDIM / 128), 256, 0, stream>>>(
      attn16, wproj16, proj_b, x1, x, kM, kDIM, kDIM, 1);

  // 6. LN2(x1) -> h16 (reuse)
  ln_kernel<<<(kM + 7) / 8, 256, 0, stream>>>(x1, n2g, n2b, h16, kM);

  // 7. fc1 out = gelu(LN2 @ Wfc1^T + b)  (f16, reuses `big`)
  gemm_kernel<<<dim3(mblocks, kHID / 128), 256, 0, stream>>>(
      h16, wfc1_16, fc1_b, big, nullptr, kM, kHID, kDIM, 2);

  // 8. out = x1 + fc1out @ Wfc2^T + b  (f32)
  gemm_kernel<<<dim3(mblocks, kDIM / 128), 256, 0, stream>>>(
      big, wfc2_16, fc2_b, out, x1, kM, kDIM, kHID, 1);
}
